// DecoderLayer_31911607009467
// MI455X (gfx1250) — matrix operand order repaired and checked by reading, not compiled, not run on hardware
//
#include <hip/hip_runtime.h>

// ---------------------------------------------------------------------------
// Transformer decoder layer on gfx1250 (MI455X). All matmuls via
// v_wmma_f32_16x16x32_bf16 (bf16 inputs, f32 accumulate). Flash-attention
// with online softmax; LDS fragment transposes synchronized with
// s_wait_dscnt (wave-private LDS regions -> no block barriers needed).
// GEMM K-loop is an explicit ping-pong (manual 2x unroll): no register
// rotation copies, two fragment load batches in flight, WMMAs overlap the
// opposite stage's global loads (partial s_wait_loadcnt).
// Workspace usage: ~220 MB of d_ws.
// ---------------------------------------------------------------------------

typedef __attribute__((ext_vector_type(16))) __bf16 v16bf;
typedef __attribute__((ext_vector_type(8)))  float  v8f;

#define DMODEL 256
#define NHEADS 8
#define HDIM   256          // key_dim per head (keras passes key_dim=d_model)
#define QKDIM  2048         // NHEADS*HDIM
#define BATCH  8
#define SEQ    1024
#define ROWS   8192         // BATCH*SEQ
#define DFF    1024

__device__ __forceinline__ unsigned short f2bf(float f) {
    unsigned u = __float_as_uint(f);
    u += 0x7FFFu + ((u >> 16) & 1u);          // round-to-nearest-even
    return (unsigned short)(u >> 16);
}

__device__ __forceinline__ v8f vzero8() {
    v8f z;
#pragma unroll
    for (int i = 0; i < 8; ++i) z[i] = 0.0f;
    return z;
}

__device__ __forceinline__ v16bf make_frag(uint4 a, uint4 b) {
    union { uint4 q[2]; v16bf v; } u;
    u.q[0] = a; u.q[1] = b;
    return u.v;
}

__device__ __forceinline__ v8f wmma_bf16(v16bf a, v16bf b, v8f c) {
    return __builtin_amdgcn_wmma_f32_16x16x32_bf16(
        /*neg_a=*/false, a, /*neg_b=*/false, b,
        /*c_mod=*/(short)0, c, /*reuse_a=*/false, /*reuse_b=*/false);
}

// Wait for this wave's outstanding LDS ops (CDNA5 split counter).
__device__ __forceinline__ void lds_fence() {
    asm volatile("s_wait_dscnt 0" ::: "memory");
}

// --------------------------- elementwise kernels ---------------------------

__global__ __launch_bounds__(256) void cvt_f32_bf16(const float* __restrict__ in,
                                                    unsigned short* __restrict__ out,
                                                    int n) {
    int i = blockIdx.x * 256 + threadIdx.x;
    if (i < n) out[i] = f2bf(in[i]);
}

// out[n*K + k] = bf16(in[k*N + n])   (weight [K,N] -> transposed bf16 [N,K])
__global__ __launch_bounds__(256) void trw_kernel(const float* __restrict__ in,
                                                  unsigned short* __restrict__ out,
                                                  int K, int N) {
    int i = blockIdx.x * 256 + threadIdx.x;
    if (i >= K * N) return;
    int k = i / N, n = i % N;
    out[(size_t)n * K + k] = f2bf(in[i]);
}

// V [b*SEQ + s][n*HDIM + h] (bf16) -> Vt [((b*NHEADS+n)*HDIM + h)*SEQ + s]
__global__ __launch_bounds__(256) void trv_kernel(const unsigned short* __restrict__ V,
                                                  unsigned short* __restrict__ Vt) {
    int i = blockIdx.x * 256 + threadIdx.x;
    if (i >= ROWS * QKDIM) return;
    int row = i >> 11;            // /2048
    int c   = i & 2047;
    int b = row >> 10, s = row & 1023;
    int n = c >> 8,    h = c & 255;
    Vt[(((size_t)(b * NHEADS + n) * HDIM) + h) * SEQ + s] = V[i];
}

// ------------------------------ GEMM kernel --------------------------------
// C[M,N] = act(A[M,K](bf16) @ W[K,N] + bias), W supplied transposed Bt[N,K].
// Block: 256 threads = 8 waves arranged 4(M) x 2(N). Wave tile: 16 x 64.
// Block tile: 64 x 128. M,N multiples of 64/128; K a multiple of 64.
// Explicit ping-pong over 32-wide K stages (manual 2x unroll): even/odd
// fragment buffers alternate, so no register rotation and each WMMA group
// depends on loads issued one full stage earlier.
__global__ __launch_bounds__(256) void gemm_bf16_kernel(
        const unsigned short* __restrict__ A,
        const unsigned short* __restrict__ Bt,
        const float* __restrict__ bias,
        float* __restrict__ outF,
        unsigned short* __restrict__ outB,
        int M, int N, int K, int relu) {
    const int lane = threadIdx.x & 31;
    const int wid  = threadIdx.x >> 5;
    const int col  = lane & 15;
    const int half = lane >> 4;
    const int m0 = blockIdx.x * 64 + (wid & 3) * 16;
    const int n0 = blockIdx.y * 128 + (wid >> 2) * 64;
    if (m0 >= M || n0 >= N) return;

    const unsigned short* arow = A + (size_t)(m0 + col) * K;

    auto loadA = [&](int k0) {
        const unsigned short* ap = arow + k0;
        return make_frag(*(const uint4*)(ap + 16 * half),
                         *(const uint4*)(ap + 16 * half + 8));
    };
    auto loadB = [&](int j, int k0) {
        const unsigned short* bp =
            Bt + (size_t)(n0 + 16 * j + col) * K + k0 + 16 * half;
        return make_frag(*(const uint4*)bp, *(const uint4*)(bp + 8));
    };

    v8f acc[4];
#pragma unroll
    for (int j = 0; j < 4; ++j) acc[j] = vzero8();

    // even-stage buffers (k0), odd-stage buffers (k0 + 32)
    v16bf aE = loadA(0);
    v16bf bE[4];
#pragma unroll
    for (int j = 0; j < 4; ++j) bE[j] = loadB(j, 0);

    for (int k0 = 0; k0 < K; k0 += 64) {
        // issue odd-stage loads, then run even-stage WMMAs
        v16bf aO = loadA(k0 + 32);                      // k0+32 < K (K % 64 == 0)
        v16bf bO[4];
#pragma unroll
        for (int j = 0; j < 4; ++j) bO[j] = loadB(j, k0 + 32);
#pragma unroll
        for (int j = 0; j < 4; ++j) acc[j] = wmma_bf16(aE, bE[j], acc[j]);

        // issue next even-stage loads (wrap: dummy prefetch on final trip)
        const int kn = (k0 + 64 < K) ? (k0 + 64) : 0;
        aE = loadA(kn);
#pragma unroll
        for (int j = 0; j < 4; ++j) bE[j] = loadB(j, kn);
#pragma unroll
        for (int j = 0; j < 4; ++j) acc[j] = wmma_bf16(aO, bO[j], acc[j]);
    }

#pragma unroll
    for (int j = 0; j < 4; ++j) {
        int c = n0 + 16 * j + col;
        float bv = bias ? bias[c] : 0.0f;
#pragma unroll
        for (int v = 0; v < 8; ++v) {
            int r = m0 + v + 8 * half;          // C layout: row = vgpr + 8*half
            float val = acc[j][v] + bv;
            if (relu) val = fmaxf(val, 0.0f);
            if (outF) outF[(size_t)r * N + c] = val;
            if (outB) outB[(size_t)r * N + c] = f2bf(val);
        }
    }
}

// --------------------------- flash attention -------------------------------
// One wave handles one (b, head, 16-row Q tile), full HDIM=256 output.
// Q,K: [b*SEQ + t][n*HDIM + h] bf16.  Vt: [((b*NH+n)*HDIM)+h][s] bf16.
// Out: same layout as Q, bf16.
__global__ __launch_bounds__(128) void attn_kernel(
        const unsigned short* __restrict__ Q,
        const unsigned short* __restrict__ Km,
        const unsigned short* __restrict__ Vt,
        unsigned short* __restrict__ Out,
        int causal) {
    __shared__ alignas(16) unsigned short qtile[4][16 * HDIM]; // 32 KB
    __shared__ float pst[4][16 * 32];                          //  8 KB

    const int lane = threadIdx.x & 31;
    const int wid  = threadIdx.x >> 5;
    const int col  = lane & 15;
    const int half = lane >> 4;

    const int gw = blockIdx.x * 4 + wid;       // 0..4095
    const int b  = gw >> 9;                    // / (NHEADS * 64)
    const int n  = (gw >> 6) & 7;
    const int t0 = (gw & 63) << 4;

    // stage this wave's 16 x 256 Q tile in LDS (wave-private region)
    const unsigned short* qbase = Q + ((size_t)(b * SEQ + t0)) * QKDIM + n * HDIM;
    unsigned short* qt = qtile[wid];
#pragma unroll
    for (int i = 0; i < 16; ++i) {
        int f = i * 32 + lane;                 // uint4 index 0..511
        int r = f >> 5, c8 = f & 31;
        *(uint4*)(qt + r * HDIM + c8 * 8) =
            *(const uint4*)(qbase + (size_t)r * QKDIM + c8 * 8);
    }
    lds_fence();

    v8f acc[16];
#pragma unroll
    for (int h = 0; h < 16; ++h) acc[h] = vzero8();
    float mst[8], lst[8];
#pragma unroll
    for (int v = 0; v < 8; ++v) { mst[v] = -3.0e38f; lst[v] = 0.0f; }

    const int sEnd = causal ? (t0 + 16) : SEQ;
    const unsigned short* vbase = Vt + (size_t)((b * NHEADS + n) * HDIM) * SEQ;

    for (int s0 = 0; s0 < sEnd; s0 += 32) {
        // ---- scores: two 16x16 tiles, K reduction over HDIM=256 ----
        const unsigned short* kcol =
            Km + (size_t)(b * SEQ + s0 + col) * QKDIM + n * HDIM + 16 * half;
        auto loadK = [&](int kc, int sub) {
            const unsigned short* p = kcol + kc * 32 + (size_t)sub * 16 * QKDIM;
            return make_frag(*(const uint4*)p, *(const uint4*)(p + 8));
        };
        auto loadQf = [&](int kc) {
            const unsigned short* qp = qt + col * HDIM + kc * 32;
            return make_frag(*(const uint4*)(qp + 16 * half),
                             *(const uint4*)(qp + 16 * half + 8));
        };

        v8f sc0 = vzero8(), sc1 = vzero8();
        v16bf k0C = loadK(0, 0), k1C = loadK(0, 1);
#pragma unroll
        for (int kc = 0; kc < 8; ++kc) {
            const int kn = (kc + 1) & 7;                // wrap: dummy prefetch
            v16bf k0N = loadK(kn, 0);
            v16bf k1N = loadK(kn, 1);
            v16bf qa  = loadQf(kc);
            sc0 = wmma_bf16(qa, k0C, sc0);
            sc1 = wmma_bf16(qa, k1C, sc1);
            k0C = k0N; k1C = k1N;
        }
        if (s0 + 32 < sEnd)
            __builtin_prefetch(Km + (size_t)(b * SEQ + s0 + 32 + col) * QKDIM + n * HDIM, 0, 0);

        // ---- online softmax over the 32 new columns ----
        float corrv[8];
#pragma unroll
        for (int v = 0; v < 8; ++v) {
            float a0 = sc0[v] * 0.0625f;       // 1/sqrt(256)
            float a1 = sc1[v] * 0.0625f;
            if (causal) {
                int tr = t0 + v + 8 * half;
                if (s0 + col > tr)      a0 = -1.0e9f;
                if (s0 + 16 + col > tr) a1 = -1.0e9f;
            }
            float mx = fmaxf(a0, a1);
#pragma unroll
            for (int o = 1; o < 16; o <<= 1) mx = fmaxf(mx, __shfl_xor(mx, o, 32));
            float mN = fmaxf(mst[v], mx);
            float corr = __expf(mst[v] - mN);
            mst[v] = mN; corrv[v] = corr;
            float p0 = __expf(a0 - mN), p1 = __expf(a1 - mN);
            float rs = p0 + p1;
#pragma unroll
            for (int o = 1; o < 16; o <<= 1) rs += __shfl_xor(rs, o, 32);
            lst[v] = lst[v] * corr + rs;
            int row = v + 8 * half;
            pst[wid][row * 32 + col]      = p0;
            pst[wid][row * 32 + 16 + col] = p1;
        }
#pragma unroll
        for (int h = 0; h < 16; ++h)
#pragma unroll
            for (int v = 0; v < 8; ++v) acc[h][v] *= corrv[v];
        lds_fence();

        // ---- P (16x32) as A fragment via LDS transpose ----
        union { unsigned short s[16]; v16bf v; } up;
        const float* pr = &pst[wid][col * 32];      // row m = lane&15
#pragma unroll
        for (int j = 0; j < 8; ++j) {
            up.s[j]     = f2bf(pr[16 * half + j]);
            up.s[8 + j] = f2bf(pr[16 * half + 8 + j]);
        }
        v16bf pA = up.v;

        // ---- acc += P @ V over all 16 output h-tiles (double-buffered) ----
        const unsigned short* vb = vbase + s0 + 16 * half;
        auto loadV = [&](int h) {
            const unsigned short* vp = vb + (size_t)(h * 16 + col) * SEQ;
            return make_frag(*(const uint4*)vp, *(const uint4*)(vp + 8));
        };
        v16bf vC = loadV(0);
#pragma unroll
        for (int h = 0; h < 16; ++h) {
            v16bf vN = loadV((h + 1) & 15);             // wrap: dummy prefetch
            acc[h] = wmma_bf16(pA, vC, acc[h]);
            vC = vN;
        }
    }

#pragma unroll
    for (int v = 0; v < 8; ++v) lst[v] = 1.0f / lst[v];
    const size_t obase = ((size_t)(b * SEQ + t0)) * QKDIM + n * HDIM;
#pragma unroll
    for (int h = 0; h < 16; ++h)
#pragma unroll
        for (int v = 0; v < 8; ++v)
            Out[obase + (size_t)(v + 8 * half) * QKDIM + h * 16 + col] =
                f2bf(acc[h][v] * lst[v]);
}

// ----------------------- residual + LayerNorm ------------------------------
// One wave per 256-element row. outF may alias resid (reads precede writes).
__global__ __launch_bounds__(256) void add_ln_kernel(
        const float* __restrict__ resid, const float* __restrict__ delta,
        const float* __restrict__ g, const float* __restrict__ be,
        float* outF, unsigned short* outB) {
    const int lane = threadIdx.x & 31;
    const int wid  = threadIdx.x >> 5;
    const int row  = blockIdx.x * 8 + wid;
    const float* rp = resid + (size_t)row * DMODEL;
    const float* dp = delta + (size_t)row * DMODEL;
    float xv[8], s = 0.0f, sq = 0.0f;
#pragma unroll
    for (int j = 0; j < 8; ++j) {
        int e = lane * 8 + j;
        float v = rp[e] + dp[e];
        xv[j] = v; s += v; sq += v * v;
    }
#pragma unroll
    for (int o = 1; o < 32; o <<= 1) { s += __shfl_xor(s, o, 32); sq += __shfl_xor(sq, o, 32); }
    float mu  = s * (1.0f / DMODEL);
    float var = sq * (1.0f / DMODEL) - mu * mu;
    float inv = rsqrtf(var + 1e-3f);
#pragma unroll
    for (int j = 0; j < 8; ++j) {
        int e = lane * 8 + j;
        float y = (xv[j] - mu) * inv * g[e] + be[e];
        outF[(size_t)row * DMODEL + e] = y;
        if (outB) outB[(size_t)row * DMODEL + e] = f2bf(y);
    }
}

// ------------------------------- launcher ----------------------------------

extern "C" void kernel_launch(void* const* d_in, const int* in_sizes, int n_in,
                              void* d_out, int out_size, void* d_ws, size_t ws_size,
                              hipStream_t stream) {
    (void)in_sizes; (void)n_in; (void)out_size; (void)ws_size;
    const float* x   = (const float*)d_in[0];
    const float* ctx = (const float*)d_in[1];
    const float *Wq1 = (const float*)d_in[2],  *bq1 = (const float*)d_in[3];
    const float *Wk1 = (const float*)d_in[4],  *bk1 = (const float*)d_in[5];
    const float *Wv1 = (const float*)d_in[6],  *bv1 = (const float*)d_in[7];
    const float *Wo1 = (const float*)d_in[8],  *bo1 = (const float*)d_in[9];
    const float *g1  = (const float*)d_in[10], *be1 = (const float*)d_in[11];
    const float *Wq2 = (const float*)d_in[12], *bq2 = (const float*)d_in[13];
    const float *Wk2 = (const float*)d_in[14], *bk2 = (const float*)d_in[15];
    const float *Wv2 = (const float*)d_in[16], *bv2 = (const float*)d_in[17];
    const float *Wo2 = (const float*)d_in[18], *bo2 = (const float*)d_in[19];
    const float *g2  = (const float*)d_in[20], *be2 = (const float*)d_in[21];
    const float *W1  = (const float*)d_in[22], *b1  = (const float*)d_in[23];
    const float *W2  = (const float*)d_in[24], *b2  = (const float*)d_in[25];
    const float *g3  = (const float*)d_in[26], *be3 = (const float*)d_in[27];

    char* ws = (char*)d_ws;
    const size_t MB = 1ull << 20;
    unsigned short* xb    = (unsigned short*)(ws + 0 * MB);    // 4 MB
    unsigned short* cb    = (unsigned short*)(ws + 4 * MB);    // 4 MB
    unsigned short* wq1t  = (unsigned short*)(ws + 8 * MB);
    unsigned short* wk1t  = (unsigned short*)(ws + 9 * MB);
    unsigned short* wv1t  = (unsigned short*)(ws + 10 * MB);
    unsigned short* wo1t  = (unsigned short*)(ws + 11 * MB);
    unsigned short* wq2t  = (unsigned short*)(ws + 12 * MB);
    unsigned short* wk2t  = (unsigned short*)(ws + 13 * MB);
    unsigned short* wv2t  = (unsigned short*)(ws + 14 * MB);
    unsigned short* wo2t  = (unsigned short*)(ws + 15 * MB);
    unsigned short* w1t   = (unsigned short*)(ws + 16 * MB);
    unsigned short* w2t   = (unsigned short*)(ws + 16 * MB + 512 * 1024);
    unsigned short* Qb    = (unsigned short*)(ws + 24 * MB);   // 32 MB
    unsigned short* Kb    = (unsigned short*)(ws + 56 * MB);   // 32 MB
    unsigned short* Vb    = (unsigned short*)(ws + 88 * MB);   // 32 MB
    unsigned short* Vtb   = (unsigned short*)(ws + 120 * MB);  // 32 MB
    unsigned short* attnb = (unsigned short*)(ws + 152 * MB);  // 32 MB
    float*          Pf    = (float*)(ws + 184 * MB);           //  8 MB
    float*          xcur  = (float*)(ws + 192 * MB);           //  8 MB
    unsigned short* xcurb = (unsigned short*)(ws + 200 * MB);  //  4 MB
    unsigned short* hb    = (unsigned short*)(ws + 204 * MB);  // 16 MB  (end 220 MB)

    const int NELEM = ROWS * DMODEL;                            // 2,097,152

    // ---- precision conversion + weight transposes ----
    cvt_f32_bf16<<<NELEM / 256, 256, 0, stream>>>(x,   xb, NELEM);
    cvt_f32_bf16<<<NELEM / 256, 256, 0, stream>>>(ctx, cb, NELEM);
    trw_kernel<<<2048, 256, 0, stream>>>(Wq1, wq1t, DMODEL, QKDIM);
    trw_kernel<<<2048, 256, 0, stream>>>(Wk1, wk1t, DMODEL, QKDIM);
    trw_kernel<<<2048, 256, 0, stream>>>(Wv1, wv1t, DMODEL, QKDIM);
    trw_kernel<<<2048, 256, 0, stream>>>(Wo1, wo1t, QKDIM, DMODEL);
    trw_kernel<<<2048, 256, 0, stream>>>(Wq2, wq2t, DMODEL, QKDIM);
    trw_kernel<<<2048, 256, 0, stream>>>(Wk2, wk2t, DMODEL, QKDIM);
    trw_kernel<<<2048, 256, 0, stream>>>(Wv2, wv2t, DMODEL, QKDIM);
    trw_kernel<<<2048, 256, 0, stream>>>(Wo2, wo2t, QKDIM, DMODEL);
    trw_kernel<<<1024, 256, 0, stream>>>(W1, w1t, DMODEL, DFF);
    trw_kernel<<<1024, 256, 0, stream>>>(W2, w2t, DFF, DMODEL);

    // ---- causal self-attention ----
    gemm_bf16_kernel<<<dim3(128, 16), 256, 0, stream>>>(xb, wq1t, bq1, nullptr, Qb, ROWS, QKDIM, DMODEL, 0);
    gemm_bf16_kernel<<<dim3(128, 16), 256, 0, stream>>>(xb, wk1t, bk1, nullptr, Kb, ROWS, QKDIM, DMODEL, 0);
    gemm_bf16_kernel<<<dim3(128, 16), 256, 0, stream>>>(xb, wv1t, bv1, nullptr, Vb, ROWS, QKDIM, DMODEL, 0);
    trv_kernel<<<(ROWS * QKDIM) / 256, 256, 0, stream>>>(Vb, Vtb);
    attn_kernel<<<1024, 128, 0, stream>>>(Qb, Kb, Vtb, attnb, 1);
    gemm_bf16_kernel<<<dim3(128, 2), 256, 0, stream>>>(attnb, wo1t, bo1, Pf, nullptr, ROWS, DMODEL, QKDIM, 0);
    add_ln_kernel<<<1024, 256, 0, stream>>>(x, Pf, g1, be1, xcur, xcurb);

    // ---- cross-attention ----
    gemm_bf16_kernel<<<dim3(128, 16), 256, 0, stream>>>(xcurb, wq2t, bq2, nullptr, Qb, ROWS, QKDIM, DMODEL, 0);
    gemm_bf16_kernel<<<dim3(128, 16), 256, 0, stream>>>(cb,    wk2t, bk2, nullptr, Kb, ROWS, QKDIM, DMODEL, 0);
    gemm_bf16_kernel<<<dim3(128, 16), 256, 0, stream>>>(cb,    wv2t, bv2, nullptr, Vb, ROWS, QKDIM, DMODEL, 0);
    trv_kernel<<<(ROWS * QKDIM) / 256, 256, 0, stream>>>(Vb, Vtb);
    attn_kernel<<<1024, 128, 0, stream>>>(Qb, Kb, Vtb, attnb, 0);
    gemm_bf16_kernel<<<dim3(128, 2), 256, 0, stream>>>(attnb, wo2t, bo2, Pf, nullptr, ROWS, DMODEL, QKDIM, 0);
    add_ln_kernel<<<1024, 256, 0, stream>>>(xcur, Pf, g2, be2, xcur, xcurb);

    // ---- FFN ----
    gemm_bf16_kernel<<<dim3(128, 8), 256, 0, stream>>>(xcurb, w1t, b1, nullptr, hb, ROWS, DFF, DMODEL, 1);
    gemm_bf16_kernel<<<dim3(128, 2), 256, 0, stream>>>(hb,    w2t, b2, Pf, nullptr, ROWS, DMODEL, DFF, 0);
    add_ln_kernel<<<1024, 256, 0, stream>>>(xcur, Pf, g3, be3, (float*)d_out, nullptr);
}
